// BatchPC_84516366451218
// MI455X (gfx1250) — matrix operand order repaired and checked by reading, not compiled, not run on hardware
//
#include <hip/hip_runtime.h>
#include <hip/hip_bf16.h>

typedef __attribute__((ext_vector_type(16))) __bf16 v16bf;
typedef __attribute__((ext_vector_type(8)))  __bf16 v8bf;
typedef __attribute__((ext_vector_type(8)))  float  v8f;

#define WMMA_BF16(A, B, C) \
    __builtin_amdgcn_wmma_f32_16x16x32_bf16(false, (A), false, (B), (short)0, (C), false, false)

// Build a v16bf operand from two 16-byte contiguous LDS runs (2x ds_load_b128)
__device__ __forceinline__ v16bf load_op(const __bf16* p0, const __bf16* p1) {
    union { v16bf v; v8bf h[2]; } u;
    u.h[0] = *(const v8bf*)p0;
    u.h[1] = *(const v8bf*)p1;
    return u.v;
}

// Convert float4 -> 4 bf16 hi + 4 bf16 lo, packed 8-byte stores (row-major dst)
__device__ __forceinline__ void cvt_store4(__bf16* hrow, __bf16* lrow, int col, float4 v) {
    union { __bf16 b[4]; unsigned long long u; } ph, pl;
    float vv[4] = {v.x, v.y, v.z, v.w};
    #pragma unroll
    for (int j = 0; j < 4; ++j) {
        __bf16 h = (__bf16)vv[j];
        ph.b[j] = h;
        pl.b[j] = (__bf16)(vv[j] - (float)h);
    }
    *(unsigned long long*)(hrow + col) = ph.u;
    *(unsigned long long*)(lrow + col) = pl.u;
}

// ---------------------------------------------------------------------------
// Kernel 0: zero the covariance accumulator region of the workspace
// ---------------------------------------------------------------------------
__global__ void zero_kernel(float* __restrict__ p, int n) {
    int i = blockIdx.x * blockDim.x + threadIdx.x;
    if (i < n) p[i] = 0.0f;
}

// ---------------------------------------------------------------------------
// Kernel 1: partial covariance  covsum += x^T x   (bf16 split-WMMA, f32 acc)
// Chunk staged TRANSPOSED (feature-major, K-minor) so both A and B operand
// gathers are contiguous ds_load_b128 pairs.
// ---------------------------------------------------------------------------
#define CPAD 40   // row stride (halves): 80 bytes, 16B-aligned, 20-bank stride

__global__ __launch_bounds__(256) void cov_kernel(const float* __restrict__ x,
                                                  float* __restrict__ covsum,
                                                  long N) {
    __shared__ __align__(16) __bf16 hiT[64][CPAD];  // [feature][k]
    __shared__ __align__(16) __bf16 loT[64][CPAD];

    const int tid  = threadIdx.x;
    const int lane = tid & 31;
    const int wave = tid >> 5;
    const int grp  = lane >> 4;   // 0: lanes 0-15, 1: lanes 16-31
    const int l15  = lane & 15;

    const long rowsPer = N / (long)gridDim.x;          // 2048
    const long base    = (long)blockIdx.x * rowsPer;
    const int  kIters  = (int)(rowsPer >> 5);          // chunks of 32 rows

    const int mi0 = wave >> 2;        // first tile-row (0..1)
    const int mi1 = mi0 + 2;          // second tile-row
    const int ni  = wave & 3;         // tile-col

    v8f acc0 = {}; v8f acc1 = {};

    for (int it = 0; it < kIters; ++it) {
        const float4* src4 = (const float4*)(x + (base + (long)it * 32) * 64);
        // 2048 floats = 512 float4; transpose-convert into hiT/loT
        for (int f = tid; f < 512; f += 256) {
            float4 v = src4[f];
            const int k   = f >> 4;          // batch row within chunk (0..31)
            const int col = (f & 15) * 4;    // feature (0..60)
            float vv[4] = {v.x, v.y, v.z, v.w};
            #pragma unroll
            for (int j = 0; j < 4; ++j) {
                __bf16 h = (__bf16)vv[j];
                hiT[col + j][k] = h;
                loT[col + j][k] = (__bf16)(vv[j] - (float)h);
            }
        }
        __syncthreads();

        // A[m][k] = x[k][mi*16+m]: row mi*16+l15, halves = K[grp*8..+7],[16+grp*8..+7]
        const __bf16* ra0 = &hiT[mi0 * 16 + l15][0];
        const __bf16* la0 = &loT[mi0 * 16 + l15][0];
        const __bf16* ra1 = &hiT[mi1 * 16 + l15][0];
        const __bf16* la1 = &loT[mi1 * 16 + l15][0];
        const __bf16* rb  = &hiT[ni  * 16 + l15][0];
        const __bf16* lb  = &loT[ni  * 16 + l15][0];

        v16bf a0h = load_op(ra0 + grp * 8, ra0 + 16 + grp * 8);
        v16bf a0l = load_op(la0 + grp * 8, la0 + 16 + grp * 8);
        v16bf a1h = load_op(ra1 + grp * 8, ra1 + 16 + grp * 8);
        v16bf a1l = load_op(la1 + grp * 8, la1 + 16 + grp * 8);
        // B[k][n]: row ni*16+l15, halves = K[grp*8..+7],[16+grp*8..+7]
        v16bf bh  = load_op(rb + grp * 8, rb + 16 + grp * 8);
        v16bf bl  = load_op(lb + grp * 8, lb + 16 + grp * 8);

        // x*x ~= hi*hi + hi*lo + lo*hi (fp32-quality via bf16 split)
        acc0 = WMMA_BF16(a0h, bh, acc0);
        acc0 = WMMA_BF16(a0h, bl, acc0);
        acc0 = WMMA_BF16(a0l, bh, acc0);
        acc1 = WMMA_BF16(a1h, bh, acc1);
        acc1 = WMMA_BF16(a1h, bl, acc1);
        acc1 = WMMA_BF16(a1l, bh, acc1);
        __syncthreads();
    }

    // C/D layout: VGPR r -> row = r + 8*grp (within tile), col = lane&15
    const int colg = ni * 16 + l15;
    #pragma unroll
    for (int r = 0; r < 8; ++r) {
        const int rloc = r + grp * 8;
        atomicAdd(&covsum[(mi0 * 16 + rloc) * 64 + colg], acc0[r]);
        atomicAdd(&covsum[(mi1 * 16 + rloc) * 64 + colg], acc1[r]);
    }
}

// ---------------------------------------------------------------------------
// Kernel 2: running-cov update + 64x64 parallel Jacobi eigendecomposition,
// top-32 selection, pivot sign fix, Q = ev / sqrt(es).  One block.
// ---------------------------------------------------------------------------
#define NSWEEP 15
__global__ __launch_bounds__(256) void eig_kernel(const float* __restrict__ covsum,
                                                  const float* __restrict__ rC,
                                                  float* __restrict__ Qout,
                                                  float invN) {
    __shared__ float A[64][65];
    __shared__ float V[64][65];
    __shared__ float cs[32], sn[32];
    __shared__ int   pp[32], qq[32];
    __shared__ float dv[64];

    const int tid = threadIdx.x;

    for (int i = tid; i < 4096; i += 256) {
        const int r = i >> 6, c = i & 63;
        const float Cb = covsum[i] * invN;
        const float rc = rC[i];
        A[r][c] = rc + 0.1f * (Cb - rc);     // MOMENTUM = 0.1
        V[r][c] = (r == c) ? 1.0f : 0.0f;
    }
    __syncthreads();

    for (int sweep = 0; sweep < NSWEEP; ++sweep) {
        for (int rr = 0; rr < 63; ++rr) {
            if (tid < 32) {
                int p, q;
                if (tid == 0) { p = 63; q = rr; }
                else { p = (rr + tid) % 63; q = (rr + 63 - tid) % 63; }
                if (p > q) { int t = p; p = q; q = t; }
                pp[tid] = p; qq[tid] = q;
                const float app = A[p][p], aqq = A[q][q], apq = A[p][q];
                float c = 1.0f, s = 0.0f;
                if (fabsf(apq) > 1e-20f) {
                    const float theta = (aqq - app) / (2.0f * apq);
                    const float t2 = copysignf(1.0f, theta) /
                                     (fabsf(theta) + sqrtf(theta * theta + 1.0f));
                    c = 1.0f / sqrtf(t2 * t2 + 1.0f);
                    s = t2 * c;
                }
                cs[tid] = c; sn[tid] = s;
            }
            __syncthreads();
            for (int i = tid; i < 2048; i += 256) {     // rows: A = J^T A
                const int k = i >> 6, j = i & 63;
                const int p = pp[k], q = qq[k];
                const float c = cs[k], s = sn[k];
                const float ap = A[p][j], aq = A[q][j];
                A[p][j] = c * ap - s * aq;
                A[q][j] = s * ap + c * aq;
            }
            __syncthreads();
            for (int i = tid; i < 2048; i += 256) {     // cols: A = A J, V = V J
                const int k = i >> 6, j = i & 63;
                const int p = pp[k], q = qq[k];
                const float c = cs[k], s = sn[k];
                const float ap = A[j][p], aq = A[j][q];
                A[j][p] = c * ap - s * aq;
                A[j][q] = s * ap + c * aq;
                const float vp = V[j][p], vq = V[j][q];
                V[j][p] = c * vp - s * vq;
                V[j][q] = s * vp + c * vq;
            }
            __syncthreads();
        }
    }

    if (tid < 64) dv[tid] = A[tid][tid];
    __syncthreads();

    if (tid < 64) {
        const float d = dv[tid];
        int rank = 0;
        for (int j = 0; j < 64; ++j) {
            const float dj = dv[j];
            if (dj > d || (dj == d && j < tid)) ++rank;
        }
        if (rank < 32) {
            float dot = 0.0f;
            for (int r = 0; r < 64; ++r)
                dot += V[r][tid] * ((float)r * (1.0f / 63.0f));
            const float sg  = (dot > 0.0f) ? 1.0f : ((dot < 0.0f) ? -1.0f : 0.0f);
            const float inv = sg / sqrtf(d);
            for (int r = 0; r < 64; ++r)
                Qout[rank * 64 + r] = V[r][tid] * inv;
        }
    }
}

// ---------------------------------------------------------------------------
// Kernel 3: y = x @ Q^T   [N,64]x[64,32] (bf16 split-WMMA, f32 out)
// x staged row-major (K-minor) -> A gathers are contiguous b128 pairs;
// Q row-major -> B gathers contiguous; B operands held in registers.
// ---------------------------------------------------------------------------
#define XPAD 72   // 144-byte row stride, 16B-aligned, 36-bank stride

__global__ __launch_bounds__(256) void gemm_kernel(const float* __restrict__ x,
                                                   const float* __restrict__ Q,
                                                   float* __restrict__ y,
                                                   long N) {
    __shared__ __align__(16) __bf16 xh[128][XPAD];
    __shared__ __align__(16) __bf16 xl[128][XPAD];
    __shared__ __align__(16) __bf16 qh[32][XPAD];
    __shared__ __align__(16) __bf16 ql[32][XPAD];

    const int tid  = threadIdx.x;
    const int lane = tid & 31;
    const int wave = tid >> 5;
    const int grp  = lane >> 4;
    const int l15  = lane & 15;

    {   // stage Q (2048 floats = 512 float4)
        const float4* q4 = (const float4*)Q;
        for (int f = tid; f < 512; f += 256)
            cvt_store4(&qh[f >> 4][0], &ql[f >> 4][0], (f & 15) * 4, q4[f]);
    }
    __syncthreads();

    // B[k][n] = Q[n][k]: row ct*16+l15, halves = K[kc*32+grp*8..],[+16..]
    v16bf Bh[2][2], Bl[2][2];
    #pragma unroll
    for (int ct = 0; ct < 2; ++ct) {
        const __bf16* rq = &qh[ct * 16 + l15][0];
        const __bf16* lq = &ql[ct * 16 + l15][0];
        #pragma unroll
        for (int kc = 0; kc < 2; ++kc) {
            Bh[ct][kc] = load_op(rq + kc * 32 + grp * 8, rq + kc * 32 + 16 + grp * 8);
            Bl[ct][kc] = load_op(lq + kc * 32 + grp * 8, lq + kc * 32 + 16 + grp * 8);
        }
    }

    const long rowsPer = N / (long)gridDim.x;   // 2048
    const long base    = (long)blockIdx.x * rowsPer;
    const int  slabs   = (int)(rowsPer >> 7);   // slabs of 128 rows

    for (int s = 0; s < slabs; ++s) {
        const float4* src4 = (const float4*)(x + (base + (long)s * 128) * 64);
        __syncthreads();
        for (int f = tid; f < 2048; f += 256)   // 8192 floats = 2048 float4
            cvt_store4(&xh[f >> 4][0], &xl[f >> 4][0], (f & 15) * 4, src4[f]);
        __syncthreads();

        // A[m][k] = x[row][k]: row wave*16+l15, halves = K[kc*32+grp*8..],[+16..]
        const __bf16* rx = &xh[wave * 16 + l15][0];
        const __bf16* lx = &xl[wave * 16 + l15][0];
        v16bf Ah[2], Al[2];
        #pragma unroll
        for (int kc = 0; kc < 2; ++kc) {
            Ah[kc] = load_op(rx + kc * 32 + grp * 8, rx + kc * 32 + 16 + grp * 8);
            Al[kc] = load_op(lx + kc * 32 + grp * 8, lx + kc * 32 + 16 + grp * 8);
        }

        const long row0 = base + (long)s * 128 + (long)wave * 16;
        #pragma unroll
        for (int ct = 0; ct < 2; ++ct) {
            v8f acc = {};
            acc = WMMA_BF16(Ah[0], Bh[ct][0], acc);
            acc = WMMA_BF16(Ah[0], Bl[ct][0], acc);
            acc = WMMA_BF16(Al[0], Bh[ct][0], acc);
            acc = WMMA_BF16(Ah[1], Bh[ct][1], acc);
            acc = WMMA_BF16(Ah[1], Bl[ct][1], acc);
            acc = WMMA_BF16(Al[1], Bh[ct][1], acc);
            #pragma unroll
            for (int r = 0; r < 8; ++r)
                y[(row0 + r + grp * 8) * 32 + ct * 16 + l15] = acc[r];
        }
    }
}

// ---------------------------------------------------------------------------
extern "C" void kernel_launch(void* const* d_in, const int* in_sizes, int n_in,
                              void* d_out, int out_size, void* d_ws, size_t ws_size,
                              hipStream_t stream) {
    (void)n_in; (void)out_size; (void)ws_size;
    const float* x  = (const float*)d_in[0];   // [N,64] f32
    const float* rC = (const float*)d_in[1];   // [64,64] f32
    float* out      = (float*)d_out;           // [N,32] f32

    const long N = (long)in_sizes[0] / 64;

    float* covsum = (float*)d_ws;              // 4096 f32
    float* Qbuf   = covsum + 4096;             // 2048 f32

    zero_kernel<<<16, 256, 0, stream>>>(covsum, 4096);
    cov_kernel<<<1024, 256, 0, stream>>>(x, covsum, N);
    eig_kernel<<<1, 256, 0, stream>>>(covsum, rC, Qbuf, 1.0f / (float)N);
    gemm_kernel<<<1024, 256, 0, stream>>>(x, Qbuf, out, N);
}
